// MuModel_78297253806629
// MI455X (gfx1250) — matrix operand order repaired and checked by reading, not compiled, not run on hardware
//
#include <hip/hip_runtime.h>

typedef __attribute__((ext_vector_type(16))) _Float16 v16h;
typedef __attribute__((ext_vector_type(8)))  float    v8f;

#define EMBED_DIM  64
#define NUM_CODES  4096
#define BETA       0.25f
#define MT         4          // row tiles (of 16 rows) per wave
#define WAVES_PER_BLOCK 8

// ---------------------------------------------------------------------------
// Kernel 1: convert codebook (64 x 4096 fp32, row-major) into WMMA-B f16
// fragments scaled by -2 (so WMMA accumulates the distance directly when the
// accumulator is seeded with ||e||^2), plus per-code squared norms.
//
// Contiguous-K B layout used here (A matches) for V_WMMA_F32_16X16X32_F16 (B is 32x16 KxN, 8 VGPRs of
// packed f16): lane = n + 16*(K/16); within a lane, element e = K%16
// (element order low/high halves of VGPR v => e = 2v+h). We store each lane's
// 16 halves contiguously (32B) so the main kernel does one v16h load per lane.
// ---------------------------------------------------------------------------
__global__ void vq_prep(const float* __restrict__ emb,
                        _Float16*    __restrict__ embH,
                        float*       __restrict__ colSq)
{
    int c = blockIdx.x * blockDim.x + threadIdx.x;
    if (c >= NUM_CODES) return;
    int ct = c >> 4;          // code tile
    int n  = c & 15;          // column within tile
    float sq = 0.f;
#pragma unroll
    for (int d = 0; d < EMBED_DIM; ++d) {
        float e = emb[d * NUM_CODES + c];
        sq += e * e;
        int kt   = d >> 5;        // which K=32 slab (0..1)
        int k    = d & 31;        // K within slab
        int hlB  = k >> 4;        // lane-half selects K block of 16
        int elem = k & 15;        // element within lane's v16h
        int lane = n + (hlB << 4);
        embH[(size_t)(((ct * 2 + kt) * 32 + lane) << 4) + elem] =
            (_Float16)(-2.0f * e);
    }
    colSq[c] = sq;
}

// ---------------------------------------------------------------------------
// Kernel 2: fused distance-GEMM (WMMA f16 -> f32) + argmin + gather + output.
// Each wave owns MT=4 tiles of 16 rows (64 rows). Streams all 256 code tiles
// with software-pipelined (double-buffered) B-fragment loads from L2.
// Distance = csq (seeded into C) + sum x * (-2e), accumulated by WMMA in f32.
// ---------------------------------------------------------------------------
__global__ __launch_bounds__(256) void vq_main(
    const float*    __restrict__ x,      // (nRows, 64) fp32
    const float*    __restrict__ emb,    // (64, 4096) fp32
    const _Float16* __restrict__ embH,   // B fragments (-2e)
    const float*    __restrict__ colSq,  // (4096) ||e||^2
    float*          __restrict__ outQ,   // (nRows, 64)
    int*            __restrict__ idxOut, // (nRows)
    float*          __restrict__ partials)
{
    const int lane   = threadIdx.x & 31;
    const int wave   = threadIdx.x >> 5;
    const int waveId = blockIdx.x * WAVES_PER_BLOCK + wave;
    const int rowBase = waveId * (16 * MT);
    const int m  = lane & 15;     // A: row within tile / B,C: column within tile
    const int hl = lane >> 4;     // lane half

    // ---- Build A fragments (16x32 f16 each; contiguous K order matching B:
    // lane half hl covers K run [hl*16, hl*16+16)) ----
    v16h a[MT][2];
#pragma unroll
    for (int t = 0; t < MT; ++t) {
        const float* row = x + (size_t)(rowBase + t * 16 + m) * EMBED_DIM;
#pragma unroll
        for (int kt = 0; kt < 2; ++kt) {
            const float* rp = row + kt * 32;
            float f[16];
            *(float4*)&f[0]  = *(const float4*)(rp + hl * 16);
            *(float4*)&f[4]  = *(const float4*)(rp + hl * 16 + 4);
            *(float4*)&f[8]  = *(const float4*)(rp + hl * 16 + 8);
            *(float4*)&f[12] = *(const float4*)(rp + hl * 16 + 12);
#pragma unroll
            for (int j = 0; j < 16; ++j) a[t][kt][j] = (_Float16)f[j];
        }
    }

    float minVal[MT][8];
    int   minIdx[MT][8];
#pragma unroll
    for (int t = 0; t < MT; ++t)
#pragma unroll
        for (int i = 0; i < 8; ++i) { minVal[t][i] = 3.402823466e38f; minIdx[t][i] = 0; }

    // ---- Stream all 4096 codes in tiles of 16 columns; double-buffer the
    // B fragments + column norms so L2 latency hides behind WMMA/VALU ----
    const v16h* bBase = (const v16h*)embH;
    v16h  b0  = bBase[lane];
    v16h  b1  = bBase[32 + lane];
    float csq = colSq[m];
#pragma unroll 2
    for (int ct = 0; ct < NUM_CODES / 16; ++ct) {
        const int nct = (ct + 1) & (NUM_CODES / 16 - 1);   // wrap: safe dummy
        v16h  nb0  = bBase[nct * 64 + lane];
        v16h  nb1  = bBase[nct * 64 + 32 + lane];
        float ncsq = colSq[nct * 16 + m];
        const int code = ct * 16 + m;

        // One csq splat per iteration serves all MT tiles (C slots of a lane
        // all belong to column n = lane%16).
        v8f cinit;
#pragma unroll
        for (int j = 0; j < 8; ++j) cinit[j] = csq;

#pragma unroll
        for (int t = 0; t < MT; ++t) {
            v8f acc = __builtin_amdgcn_wmma_f32_16x16x32_f16(
                false, a[t][0], false, b0, (short)0, cinit, false, false);
            acc = __builtin_amdgcn_wmma_f32_16x16x32_f16(
                false, a[t][1], false, b1, (short)0, acc, false, false);
#pragma unroll
            for (int i = 0; i < 8; ++i) {
                // acc[i] == ||e||^2 - 2 x.e  (row-constant ||x||^2 omitted)
                float d = acc[i];
                // one shared v_cmp + two v_cndmask
                bool lt = d < minVal[t][i];
                minIdx[t][i] = lt ? code : minIdx[t][i];
                minVal[t][i] = lt ? d    : minVal[t][i];
            }
        }
        b0 = nb0; b1 = nb1; csq = ncsq;
    }

    // ---- Cross-lane argmin within each 16-lane half (C layout: slot i holds
    // row M = i + 8*hl, column N = lane%16). Tie-break on smaller index to
    // match argmax-first-occurrence semantics. ----
#pragma unroll
    for (int t = 0; t < MT; ++t)
#pragma unroll
        for (int i = 0; i < 8; ++i)
#pragma unroll
            for (int off = 8; off >= 1; off >>= 1) {
                float ov = __shfl_xor(minVal[t][i], off, 32);
                int   oi = __shfl_xor(minIdx[t][i], off, 32);
                if (ov < minVal[t][i] ||
                    (ov == minVal[t][i] && oi < minIdx[t][i])) {
                    minVal[t][i] = ov; minIdx[t][i] = oi;
                }
            }

    // ---- Emit idx, gather winning codes (fp32 codebook), write straight-
    // through output (== quantized numerically), accumulate loss partial ----
    float lsum = 0.f;
#pragma unroll
    for (int t = 0; t < MT; ++t) {
        if (m == 0) {
#pragma unroll
            for (int i = 0; i < 8; ++i)
                idxOut[rowBase + t * 16 + hl * 8 + i] = minIdx[t][i];
        }
#pragma unroll
        for (int i = 0; i < 8; ++i) {
            int r    = rowBase + t * 16 + hl * 8 + i;
            int bidx = minIdx[t][i];       // all lanes in half agree
            const float* xr = x + (size_t)r * EMBED_DIM + m * 4;
            float4 xv = *(const float4*)xr;
            float4 q;
            q.x = emb[(m * 4 + 0) * NUM_CODES + bidx];
            q.y = emb[(m * 4 + 1) * NUM_CODES + bidx];
            q.z = emb[(m * 4 + 2) * NUM_CODES + bidx];
            q.w = emb[(m * 4 + 3) * NUM_CODES + bidx];
            *(float4*)(outQ + (size_t)r * EMBED_DIM + m * 4) = q;
            float dx = q.x - xv.x, dy = q.y - xv.y;
            float dz = q.z - xv.z, dw = q.w - xv.w;
            lsum = __builtin_fmaf(dx, dx, lsum);
            lsum = __builtin_fmaf(dy, dy, lsum);
            lsum = __builtin_fmaf(dz, dz, lsum);
            lsum = __builtin_fmaf(dw, dw, lsum);
        }
    }
#pragma unroll
    for (int off = 16; off >= 1; off >>= 1) lsum += __shfl_xor(lsum, off, 32);
    if (lane == 0) partials[waveId] = lsum;   // no atomics -> deterministic
}

// ---------------------------------------------------------------------------
// Kernel 3: deterministic fixed-order reduction of wave partials -> loss.
// ---------------------------------------------------------------------------
__global__ void vq_finalize(const float* __restrict__ partials, int n,
                            float* __restrict__ lossOut, float invTotal)
{
    int lane = threadIdx.x;
    float s = 0.f;
    for (int i = lane; i < n; i += 32) s += partials[i];
#pragma unroll
    for (int off = 16; off >= 1; off >>= 1) s += __shfl_xor(s, off, 32);
    if (lane == 0) *lossOut = BETA * s * invTotal;
}

// ---------------------------------------------------------------------------
extern "C" void kernel_launch(void* const* d_in, const int* in_sizes, int n_in,
                              void* d_out, int out_size, void* d_ws, size_t ws_size,
                              hipStream_t stream)
{
    const float* x   = (const float*)d_in[0];   // (512,8192,1) fp32
    const float* emb = (const float*)d_in[1];   // (64,4096) fp32

    const int nElem = in_sizes[0];              // 4194304
    const int nRows = nElem / EMBED_DIM;        // 65536

    // d_out layout (tuple flattened): [quantized: nElem f32][loss: 1 f32]
    // [idx: nRows int32 written through the float buffer]
    float* outQ    = (float*)d_out;
    float* lossOut = outQ + nElem;
    int*   idxOut  = (int*)(outQ + nElem + 1);

    // workspace layout
    char*     ws       = (char*)d_ws;
    _Float16* embH     = (_Float16*)ws;                                   // 512 KB
    float*    colSq    = (float*)(ws + (size_t)NUM_CODES * EMBED_DIM * 2);// 16 KB
    float*    partials = colSq + NUM_CODES;                               // nWaves f32

    const int nWaves  = nRows / (16 * MT);            // 1024
    const int nBlocks = nWaves / WAVES_PER_BLOCK;     // 128

    vq_prep<<<NUM_CODES / 256, 256, 0, stream>>>(emb, embH, colSq);
    vq_main<<<nBlocks, 32 * WAVES_PER_BLOCK, 0, stream>>>(
        x, emb, embH, colSq, outQ, idxOut, partials);
    vq_finalize<<<1, 32, 0, stream>>>(partials, nWaves, lossOut,
                                      1.0f / (float)nElem);
}
